// Attention_89550068122402
// MI455X (gfx1250) — matrix operand order repaired and checked by reading, not compiled, not run on hardware
//
#include <hip/hip_runtime.h>
#include <hip/hip_bf16.h>
#include <math.h>
#include <stdint.h>

#define BATCH 8
#define SQ 1024
#define SKL 1024
#define DMODEL 768
#define NHEAD 12
#define HDIM 64

typedef __attribute__((ext_vector_type(16))) _Float16 v16h;
typedef __attribute__((ext_vector_type(8)))  _Float16 v8h;
typedef __attribute__((ext_vector_type(8)))  float    v8f;
typedef __attribute__((ext_vector_type(4)))  float    v4f;

// ---------- fragment helpers (wave32 WMMA layouts, ISA 7.12.2) ----------

// B fragment: 16 contiguous halves (col = lane&15, K run = 16*(lane>>4))
__device__ __forceinline__ v16h load_frag_b16(const _Float16* p) {
  v8h lo = *(const v8h*)p;
  v8h hi = *(const v8h*)(p + 8);
  v16h r;
#pragma unroll
  for (int i = 0; i < 8; ++i) { r[i] = lo[i]; r[8 + i] = hi[i]; }
  return r;
}

// A fragment: two contiguous 8-half runs (row = lane&15; runs at K = 16*lh and 16*lh+8)
__device__ __forceinline__ v16h load_frag_a16(const _Float16* run0, const _Float16* run1) {
  v8h lo = *(const v8h*)run0;
  v8h hi = *(const v8h*)run1;
  v16h r;
#pragma unroll
  for (int i = 0; i < 8; ++i) { r[i] = lo[i]; r[8 + i] = hi[i]; }
  return r;
}

// A fragment built from f32 source (two runs of 8 floats), converted to f16
__device__ __forceinline__ v16h cvt_frag_a_f32(const float* run0, const float* run1) {
  v4f a0 = ((const v4f*)run0)[0];
  v4f a1 = ((const v4f*)run0)[1];
  v4f b0 = ((const v4f*)run1)[0];
  v4f b1 = ((const v4f*)run1)[1];
  v16h r;
#pragma unroll
  for (int i = 0; i < 4; ++i) {
    r[i]      = (_Float16)a0[i];
    r[4 + i]  = (_Float16)a1[i];
    r[8 + i]  = (_Float16)b0[i];
    r[12 + i] = (_Float16)b1[i];
  }
  return r;
}

__device__ __forceinline__ v8f wmma_f16(v16h a, v16h b, v8f c) {
  return __builtin_amdgcn_wmma_f32_16x16x32_f16(false, a, false, b, (short)0, c,
                                                false, false);
}

__device__ __forceinline__ float hmax16(float v) {
#pragma unroll
  for (int m = 1; m < 16; m <<= 1) v = fmaxf(v, __shfl_xor(v, m, 32));
  return v;
}
__device__ __forceinline__ float hsum16(float v) {
#pragma unroll
  for (int m = 1; m < 16; m <<= 1) v += __shfl_xor(v, m, 32);
  return v;
}

// Async copy 16B global -> LDS (gfx1250 GLOBAL_LOAD_ASYNC_TO_LDS_B128, ASYNCcnt).
// Generic shared pointer low 32 bits == LDS byte offset (LDS aperture layout).
__device__ __forceinline__ void async_load_b128(const _Float16* gaddr, _Float16* lds) {
  uint32_t loff = (uint32_t)(uintptr_t)lds;
  asm volatile("global_load_async_to_lds_b128 %0, %1, off"
               :: "v"(loff), "v"(gaddr)
               : "memory");
}

// ---------- kernel 1: transpose + convert weights to f16 [N][K] ----------
__global__ void wtrans_kernel(const float* __restrict__ Wq,
                              const float* __restrict__ Wk,
                              const float* __restrict__ Wv,
                              _Float16* __restrict__ WT) {
  int idx = blockIdx.x * blockDim.x + threadIdx.x;
  const int DD = DMODEL * DMODEL;
  if (idx >= 3 * DD) return;
  int w = idx / DD;
  int r = idx - w * DD;
  int n = r / DMODEL;
  int k = r - n * DMODEL;
  const float* src = (w == 0) ? Wq : (w == 1) ? Wk : Wv;
  WT[(size_t)w * DD + (size_t)n * DMODEL + k] = (_Float16)src[(size_t)k * DMODEL + n];
}

// ---------- kernel 2: fused QKV projection GEMMs (software-pipelined) ----------
__global__ void proj_kernel(const float* __restrict__ hidden,
                            const float* __restrict__ context,
                            const _Float16* __restrict__ WT,   // 3 stacked [N][K]
                            const float* __restrict__ bq,
                            const float* __restrict__ bk,
                            const float* __restrict__ bv,
                            _Float16* __restrict__ Qb,
                            _Float16* __restrict__ Kb,
                            _Float16* __restrict__ Vt) {
  const int wave = threadIdx.x >> 5;
  const int lane = threadIdx.x & 31;
  const int lh = lane >> 4;
  const int ll = lane & 15;

  const int which = blockIdx.y;
  const int unit = blockIdx.x * 8 + wave;          // 0..6143
  const int tn = unit % (DMODEL / 64);             // 12 n-tiles of 64
  const int tm = unit / (DMODEL / 64);             // 512 m-tiles of 16
  const int mbase = tm * 16;
  const int nbase = tn * 64;

  const float* A = (which == 0) ? hidden : context;
  const _Float16* W = WT + (size_t)which * DMODEL * DMODEL;
  const float* bias = (which == 0) ? bq : (which == 1) ? bk : bv;

  const float* arow = A + (size_t)(mbase + ll) * DMODEL;
  const _Float16* wrow = W + (size_t)(nbase + ll) * DMODEL + lh * 16;

  // prologue: fragments for k0 = 0
  v16h a  = cvt_frag_a_f32(arow + lh * 16, arow + lh * 16 + 8);
  v16h b0 = load_frag_b16(wrow);
  v16h b1 = load_frag_b16(wrow + (size_t)16 * DMODEL);
  v16h b2 = load_frag_b16(wrow + (size_t)32 * DMODEL);
  v16h b3 = load_frag_b16(wrow + (size_t)48 * DMODEL);

  v8f c0 = {}, c1 = {}, c2 = {}, c3 = {};
  for (int k0 = 0; k0 < DMODEL; k0 += 32) {
    // issue next k-step's loads first (consumed next iteration)
    const int kn = (k0 + 32 < DMODEL) ? (k0 + 32) : 0;   // last iter: dummy reload
    v16h an  = cvt_frag_a_f32(arow + kn + lh * 16, arow + kn + lh * 16 + 8);
    v16h nb0 = load_frag_b16(wrow + kn);
    v16h nb1 = load_frag_b16(wrow + kn + (size_t)16 * DMODEL);
    v16h nb2 = load_frag_b16(wrow + kn + (size_t)32 * DMODEL);
    v16h nb3 = load_frag_b16(wrow + kn + (size_t)48 * DMODEL);
    // compute on previously loaded fragments
    c0 = wmma_f16(a, b0, c0);
    c1 = wmma_f16(a, b1, c1);
    c2 = wmma_f16(a, b2, c2);
    c3 = wmma_f16(a, b3, c3);
    a = an; b0 = nb0; b1 = nb1; b2 = nb2; b3 = nb3;
  }

  v8f acc[4] = {c0, c1, c2, c3};
  _Float16* OutQK = (which == 0) ? Qb : Kb;
#pragma unroll
  for (int t = 0; t < 4; ++t) {
    const int col = nbase + t * 16 + ll;
    const float bv_ = bias[col];
    const int m0 = mbase + lh * 8;
    if (which != 2) {
#pragma unroll
      for (int i = 0; i < 8; ++i)
        OutQK[(size_t)(m0 + i) * DMODEL + col] = (_Float16)(acc[t][i] + bv_);
    } else {
      const int bb = m0 / SKL;
      const int sk = m0 - bb * SKL;
      v8h pk;
#pragma unroll
      for (int i = 0; i < 8; ++i) pk[i] = (_Float16)(acc[t][i] + bv_);
      *(v8h*)(Vt + ((size_t)bb * DMODEL + col) * SKL + sk) = pk;
    }
  }
}

// ---------- kernel 3: flash attention, block-shared K/V via async LDS ----------
// Block = 8 waves sharing one (b,h), covering 128 q rows. K/V tiles are staged
// in double-buffered LDS with global_load_async_to_lds_b128 and shared by all waves.
__global__ void attn_kernel(const _Float16* __restrict__ Qb,   // [B*SQ][D]
                            const _Float16* __restrict__ Kb,   // [B*SK][D]
                            const _Float16* __restrict__ Vt,   // [B][D][SK]
                            const float* __restrict__ mask,    // [B][SK]
                            float* __restrict__ out) {         // [B][SQ][D]
  __shared__ _Float16 ldsK[2][32 * 64];   // [sk][hd] tile, 4 KB per buffer
  __shared__ _Float16 ldsV[2][64 * 32];   // [hd][sk] tile, 4 KB per buffer
  __shared__ _Float16 ldsP[8][16 * 32];   // per-wave P staging

  const int tid  = threadIdx.x;
  const int wave = tid >> 5;
  const int lane = tid & 31;
  const int lh = lane >> 4;
  const int ll = lane & 15;

  const int bh   = blockIdx.x >> 3;       // 0..95, shared by whole block
  const int qblk = blockIdx.x & 7;
  const int b = bh / NHEAD;
  const int h = bh - b * NHEAD;
  const int q0 = (qblk * 8 + wave) * 16;  // this wave's 16 q rows

  // Q fragments for full HD=64, kept in registers
  const _Float16* qrow = Qb + ((size_t)b * SQ + q0 + ll) * DMODEL + h * HDIM;
  const v16h qa0 = load_frag_a16(qrow + lh * 16,      qrow + lh * 16 + 8);
  const v16h qa1 = load_frag_a16(qrow + 32 + lh * 16, qrow + 32 + lh * 16 + 8);

  const _Float16* kbase = Kb + (size_t)b * SKL * DMODEL + h * HDIM;
  const _Float16* vbase = Vt + ((size_t)b * DMODEL + h * HDIM) * SKL;
  const float* mp = mask + (size_t)b * SKL;
  _Float16* myp = &ldsP[wave][0];

  // cooperative async tile-load coordinates (one b128 per thread per tile)
  const int kr = tid >> 3, kc = (tid & 7) * 8;   // K tile: 32 rows x 64 halves
  const int vr = tid >> 2, vc = (tid & 3) * 8;   // V tile: 64 rows x 32 halves

  // prologue: issue async loads for tile 0
  async_load_b128(kbase + (size_t)kr * DMODEL + kc, &ldsK[0][kr * 64 + kc]);
  async_load_b128(vbase + (size_t)vr * SKL + vc,    &ldsV[0][vr * 32 + vc]);

  v8f o0 = {}, o1 = {}, o2 = {}, o3 = {};
  v8f mrow, lrow = {};
#pragma unroll
  for (int i = 0; i < 8; ++i) mrow[i] = -1e30f;

  int cur = 0;
  for (int s0 = 0; s0 < SKL; s0 += 32, cur ^= 1) {
    // current tile arrived (my share) + everyone's share / everyone done with
    // the buffer we are about to overwrite
    asm volatile("s_wait_asynccnt 0" ::: "memory");
    __syncthreads();

    // issue async loads for the NEXT tile into the other buffer; DMA overlaps
    // the entire compute body below
    if (s0 + 32 < SKL) {
      async_load_b128(kbase + (size_t)(s0 + 32 + kr) * DMODEL + kc,
                      &ldsK[cur ^ 1][kr * 64 + kc]);
      async_load_b128(vbase + (size_t)vr * SKL + (s0 + 32) + vc,
                      &ldsV[cur ^ 1][vr * 32 + vc]);
    }

    const _Float16* Ktile = &ldsK[cur][0];
    const _Float16* Vtile = &ldsV[cur][0];

    // ---- scores: two 16x16 tiles, S = Q*K^T/8 + mask (K frags from LDS) ----
    v8f s[2];
#pragma unroll
    for (int t = 0; t < 2; ++t) {
      const _Float16* krow = Ktile + (t * 16 + ll) * 64;
      v16h kb0 = load_frag_b16(krow + lh * 16);
      v16h kb1 = load_frag_b16(krow + 32 + lh * 16);
      v8f a = {};
      a = wmma_f16(qa0, kb0, a);
      a = wmma_f16(qa1, kb1, a);
      const float mk = mp[s0 + t * 16 + ll];
#pragma unroll
      for (int i = 0; i < 8; ++i) s[t][i] = a[i] * 0.125f + mk;
    }

    // ---- online softmax (rows live in 16-lane halves -> xor-shuffle reduce) ----
    v8f tmax;
#pragma unroll
    for (int i = 0; i < 8; ++i) tmax[i] = hmax16(fmaxf(s[0][i], s[1][i]));
    v8f mnew, alpha;
#pragma unroll
    for (int i = 0; i < 8; ++i) {
      mnew[i] = fmaxf(mrow[i], tmax[i]);
      alpha[i] = __expf(mrow[i] - mnew[i]);
    }
    mrow = mnew;

    v8f psum = {};
#pragma unroll
    for (int t = 0; t < 2; ++t)
#pragma unroll
      for (int i = 0; i < 8; ++i) {
        const float p = __expf(s[t][i] - mnew[i]);
        psum[i] += p;
        myp[(lh * 8 + i) * 32 + t * 16 + ll] = (_Float16)p;   // LDS: P[row][col]
      }
#pragma unroll
    for (int i = 0; i < 8; ++i) {
      lrow[i] = lrow[i] * alpha[i] + hsum16(psum[i]);
      o0[i] *= alpha[i]; o1[i] *= alpha[i]; o2[i] *= alpha[i]; o3[i] *= alpha[i];
    }

    // ensure P is in LDS before re-reading it as an A fragment (same wave)
    asm volatile("s_wait_dscnt 0" ::: "memory");
    v16h pa = load_frag_a16(myp + ll * 32 + lh * 16, myp + ll * 32 + lh * 16 + 8);

    // ---- O += P * V (V frags from LDS) ----
    const _Float16* vt0 = Vtile + ll * 32 + lh * 16;
    v16h vb0 = load_frag_b16(vt0);
    v16h vb1 = load_frag_b16(vt0 + 16 * 32);
    v16h vb2 = load_frag_b16(vt0 + 32 * 32);
    v16h vb3 = load_frag_b16(vt0 + 48 * 32);
    o0 = wmma_f16(pa, vb0, o0);
    o1 = wmma_f16(pa, vb1, o1);
    o2 = wmma_f16(pa, vb2, o2);
    o3 = wmma_f16(pa, vb3, o3);
  }

  // ---- epilogue: out[b][q][h*64+hd] = O * (1/l) ----
  v8f rinv;
#pragma unroll
  for (int i = 0; i < 8; ++i) rinv[i] = 1.0f / lrow[i];
  float* op = out + ((size_t)b * SQ + q0) * DMODEL + h * HDIM;
  v8f oacc[4] = {o0, o1, o2, o3};
#pragma unroll
  for (int n = 0; n < 4; ++n)
#pragma unroll
    for (int i = 0; i < 8; ++i)
      op[(size_t)(lh * 8 + i) * DMODEL + n * 16 + ll] = oacc[n][i] * rinv[i];
}

extern "C" void kernel_launch(void* const* d_in, const int* in_sizes, int n_in,
                              void* d_out, int out_size, void* d_ws, size_t ws_size,
                              hipStream_t stream) {
  const float* hidden  = (const float*)d_in[0];
  const float* context = (const float*)d_in[1];
  const float* mask    = (const float*)d_in[2];
  const float* Wq      = (const float*)d_in[3];
  const float* bq      = (const float*)d_in[4];
  const float* Wk      = (const float*)d_in[5];
  const float* bk      = (const float*)d_in[6];
  const float* Wv      = (const float*)d_in[7];
  const float* bv      = (const float*)d_in[8];
  float* out = (float*)d_out;

  const size_t DD = (size_t)DMODEL * DMODEL;
  const size_t SD = (size_t)BATCH * SQ * DMODEL;

  _Float16* WT = (_Float16*)d_ws;        // 3 * D*D
  _Float16* Qb = WT + 3 * DD;            // B*SQ*D
  _Float16* Kb = Qb + SD;                // B*SK*D
  _Float16* Vt = Kb + SD;                // B*D*SK (transposed V)

  // 1) weight transpose+convert
  {
    const int total = 3 * DMODEL * DMODEL;
    wtrans_kernel<<<(total + 255) / 256, 256, 0, stream>>>(Wq, Wk, Wv, WT);
  }
  // 2) fused QKV projections: 512 m-tiles * 12 n-tiles / 8 waves = 768 blocks, y = {Q,K,V}
  {
    dim3 grid(768, 3, 1);
    proj_kernel<<<grid, 256, 0, stream>>>(hidden, context, WT, bq, bk, bv, Qb, Kb, Vt);
  }
  // 3) attention: 96 (b,h) blocks x 8 q-blocks of 128 rows = 768 blocks
  {
    attn_kernel<<<768, 256, 0, stream>>>(Qb, Kb, Vt, mask, out);
  }
}
